// TreeEncoder_W2V_69011534512644
// MI455X (gfx1250) — compile-verified, matrix operand order checked by reading, not run on hardware
//
#include <hip/hip_runtime.h>
#include <hip/hip_bf16.h>

typedef __attribute__((ext_vector_type(16))) _Float16 v16h;
typedef __attribute__((ext_vector_type(8)))  _Float16 v8h;
typedef __attribute__((ext_vector_type(8)))  float    v8f;

union AFrag { v16h v; v8h c2[2]; _Float16 h[16]; };

// ISA 16-bit A/B fragment K mapping (CDNA5 ISA, 16-bit A-Matrix 16x32):
// lanes 0-15 : halfs 0..7 -> K=0..7,  halfs 8..15 -> K=16..23
// lanes 16-31: halfs 0..7 -> K=8..15, halfs 8..15 -> K=24..31
__device__ __forceinline__ int kmap(int lane, int hh) {
    int grp = (lane >= 16) ? 8 : 0;
    return (hh < 8) ? (hh + grp) : (hh + 8 + grp);
}

__device__ __forceinline__ float sigmoidf(float v) {
    return 1.0f / (1.0f + __expf(-v));
}

// ---------------------------------------------------------------------------
// Weight pre-swizzle: pack B = [Wcx | Wox] (256x512) into WMMA B-fragment
// order: frag[kt*NT+nt][lane][16 halfs], one contiguous 32B v16h per lane.
// ---------------------------------------------------------------------------
__global__ void k_swz_leaf(const float* __restrict__ Wcx,
                           const float* __restrict__ Wox,
                           _Float16* __restrict__ out) {
    int id = blockIdx.x * 256 + threadIdx.x;          // 8 kt * 32 nt * 32 * 16
    if (id >= 8 * 32 * 32 * 16) return;
    int hh   = id & 15;
    int lane = (id >> 4) & 31;
    int t    = id >> 9;          // kt*32 + nt
    int nt   = t & 31;
    int kt   = t >> 5;
    int k = kt * 32 + kmap(lane, hh);
    int n = nt * 16 + (lane & 15);
    float v = (n < 256) ? Wcx[k * 256 + n] : Wox[k * 256 + (n - 256)];
    out[id] = (_Float16)v;
}

// B = vstack(Wl, Wr) : 512 x 1024
__global__ void k_swz_tree(const float* __restrict__ Wl,
                           const float* __restrict__ Wr,
                           _Float16* __restrict__ out) {
    int id = blockIdx.x * 256 + threadIdx.x;          // 16 kt * 64 nt * 32 * 16
    if (id >= 16 * 64 * 32 * 16) return;
    int hh   = id & 15;
    int lane = (id >> 4) & 31;
    int t    = id >> 9;          // kt*64 + nt
    int nt   = t & 63;
    int kt   = t >> 6;
    int k = kt * 32 + kmap(lane, hh);
    int n = nt * 16 + (lane & 15);
    float v = (k < 256) ? Wl[k * 1024 + n] : Wr[(k - 256) * 1024 + n];
    out[id] = (_Float16)v;
}

// ---------------------------------------------------------------------------
// Leaf kernel: g = x(131072x256) @ [Wcx|Wox](256x512); then
//   c = g[:,0:256]+bcx ; h = sigmoid(g[:,256:512]+box) * tanh(c)
// Block: 16 rows x 512 cols. 8 waves, each owns 64 cols (4 n-tiles), K=256.
// x is f32 (read exactly once) -> in-register cvt to f16 fragments.
// Writes h as f32 (d_out) and f16 (level-1 A matrix), c as f32.
// ---------------------------------------------------------------------------
__global__ __launch_bounds__(256) void
k_leaf(const float* __restrict__ x, const _Float16* __restrict__ Bf,
       const float* __restrict__ bcx, const float* __restrict__ box,
       float* __restrict__ h_out, _Float16* __restrict__ h16_out,
       float* __restrict__ c_out) {
    __shared__ float gs[16][512 + 8];
    const int tid  = threadIdx.x;
    const int lane = tid & 31;
    const int wave = tid >> 5;
    const int mblk = blockIdx.x;
    const int row  = mblk * 16 + (lane & 15);   // M rows are exact multiple of 16
    const int grp8 = (lane >> 4) * 8;
    const float* arow = x + (size_t)row * 256;

    const v8f vzero = {0.f, 0.f, 0.f, 0.f, 0.f, 0.f, 0.f, 0.f};
    v8f acc[4];
#pragma unroll
    for (int t = 0; t < 4; ++t) acc[t] = vzero;

#pragma unroll
    for (int ks = 0; ks < 8; ++ks) {
        __builtin_prefetch(arow + ks * 32 + 96, 0, 1);  // global_prefetch_b8
        AFrag a;
        const int k0 = ks * 32 + grp8;
#pragma unroll
        for (int i = 0; i < 8; ++i) a.h[i]     = (_Float16)arow[k0 + i];
#pragma unroll
        for (int i = 0; i < 8; ++i) a.h[8 + i] = (_Float16)arow[k0 + 16 + i];
#pragma unroll
        for (int t = 0; t < 4; ++t) {
            const int ntg = wave * 4 + t;
            const v16h* bp =
                (const v16h*)(Bf + (((size_t)(ks * 32 + ntg)) * 32 + lane) * 16);
            acc[t] = __builtin_amdgcn_wmma_f32_16x16x32_f16(
                false, a.v, false, *bp, (short)0, acc[t], false, false);
        }
    }

    // D layout: lane l, vgpr r -> m = r + 8*(l>=16), n = l%16
    const int dm = grp8;
    const int dn = lane & 15;
#pragma unroll
    for (int t = 0; t < 4; ++t) {
        const int nb = wave * 64 + t * 16 + dn;
#pragma unroll
        for (int r = 0; r < 8; ++r) gs[dm + r][nb] = acc[t][r];
    }
    __syncthreads();

    const int j  = tid;                  // 0..255 output column
    const float bc = bcx[j];
    const float bo = box[j];
#pragma unroll 4
    for (int r = 0; r < 16; ++r) {
        const size_t grow = (size_t)mblk * 16 + r;
        float c = gs[r][j] + bc;
        float o = sigmoidf(gs[r][256 + j] + bo);
        float h = o * tanhf(c);
        c_out[grow * 256 + j]   = c;
        h_out[grow * 256 + j]   = h;
        h16_out[grow * 256 + j] = (_Float16)h;
    }
}

// ---------------------------------------------------------------------------
// Tree level kernel: A = h_in (f16) viewed as n_out x 512 (rows 2n,2n+1
// concat), B = vstack(Wl,Wr) 512x1024 (pre-swizzled f16 fragments).
// A fragment = two 16B v8h loads per lane, no conversion in the hot loop.
// Block: 16 rows x 1024 cols. 8 waves x 128 cols (8 n-tiles), K=512.
// K-loop fully unrolled so the scheduler can hoist next-step B-fragment
// loads above the current step's WMMAs (keeps the XDL pipe fed).
// ---------------------------------------------------------------------------
__global__ __launch_bounds__(256) void
k_level(const _Float16* __restrict__ h_in, const float* __restrict__ c_in,
        const _Float16* __restrict__ Bf,
        const float* __restrict__ bl, const float* __restrict__ br,
        float* __restrict__ c_out, _Float16* __restrict__ h16_out,
        float* __restrict__ h32_out, int n_out) {
    __shared__ float gs[16][1024 + 8];
    const int tid  = threadIdx.x;
    const int lane = tid & 31;
    const int wave = tid >> 5;
    const int mblk = blockIdx.x;
    const int row  = mblk * 16 + (lane & 15);
    const bool rv  = row < n_out;
    const int rowc = rv ? row : 0;                 // clamp: row 0 always valid
    const _Float16* arow = h_in + (size_t)rowc * 512;
    const int grp8 = (lane >> 4) * 8;
    const v8h hzero = {(_Float16)0.f, (_Float16)0.f, (_Float16)0.f, (_Float16)0.f,
                       (_Float16)0.f, (_Float16)0.f, (_Float16)0.f, (_Float16)0.f};

    const v8f vzero = {0.f, 0.f, 0.f, 0.f, 0.f, 0.f, 0.f, 0.f};
    v8f acc[8];
#pragma unroll
    for (int t = 0; t < 8; ++t) acc[t] = vzero;

#pragma unroll
    for (int ks = 0; ks < 16; ++ks) {
        __builtin_prefetch(arow + ks * 32 + 128, 0, 1);  // global_prefetch_b8
        AFrag a;
        const int k0 = ks * 32 + grp8;
        v8h lo = *(const v8h*)(arow + k0);        // halfs 0..7   (16B aligned)
        v8h hi = *(const v8h*)(arow + k0 + 16);   // halfs 8..15
        a.c2[0] = rv ? lo : hzero;
        a.c2[1] = rv ? hi : hzero;
#pragma unroll
        for (int t = 0; t < 8; ++t) {
            const int ntg = wave * 8 + t;
            const v16h* bp =
                (const v16h*)(Bf + (((size_t)(ks * 64 + ntg)) * 32 + lane) * 16);
            acc[t] = __builtin_amdgcn_wmma_f32_16x16x32_f16(
                false, a.v, false, *bp, (short)0, acc[t], false, false);
        }
    }

    const int dm = grp8;
    const int dn = lane & 15;
#pragma unroll
    for (int t = 0; t < 8; ++t) {
        const int nb = wave * 128 + t * 16 + dn;
#pragma unroll
        for (int r = 0; r < 8; ++r) gs[dm + r][nb] = acc[t][r];
    }
    __syncthreads();

    const int j = tid;                     // 0..255 memory-dim column
    const float b_i  = bl[j]       + br[j];
    const float b_lf = bl[256 + j] + br[256 + j];
    const float b_rf = bl[512 + j] + br[512 + j];
    const float b_u  = bl[768 + j] + br[768 + j];
    int rmax = n_out - mblk * 16;
    if (rmax > 16) rmax = 16;
    for (int r = 0; r < rmax; ++r) {
        const size_t grow = (size_t)mblk * 16 + r;
        const float i_  = sigmoidf(gs[r][j]       + b_i);
        const float lf  = sigmoidf(gs[r][256 + j] + b_lf);
        const float rf  = sigmoidf(gs[r][512 + j] + b_rf);
        const float u   = tanhf  (gs[r][768 + j] + b_u);
        const float lc  = c_in[(grow * 2)     * 256 + j];
        const float rc  = c_in[(grow * 2 + 1) * 256 + j];
        const float c   = i_ * u + lf * lc + rf * rc;
        const float h   = tanhf(c);
        c_out[grow * 256 + j]   = c;
        h16_out[grow * 256 + j] = (_Float16)h;
        if (h32_out) h32_out[grow * 256 + j] = h;   // root only
    }
}

// ---------------------------------------------------------------------------
// Host launcher
// ---------------------------------------------------------------------------
extern "C" void kernel_launch(void* const* d_in, const int* in_sizes, int n_in,
                              void* d_out, int out_size, void* d_ws, size_t ws_size,
                              hipStream_t stream) {
    (void)in_sizes; (void)n_in; (void)out_size; (void)ws_size;
    const float* x   = (const float*)d_in[0];
    const float* Wcx = (const float*)d_in[1];
    const float* bcx = (const float*)d_in[2];
    const float* Wox = (const float*)d_in[3];
    const float* box = (const float*)d_in[4];
    const float* Wl  = (const float*)d_in[5];
    const float* bl  = (const float*)d_in[6];
    const float* Wr  = (const float*)d_in[7];
    const float* br  = (const float*)d_in[8];
    float* out = (float*)d_out;

    const size_t N = 131072, M = 256;

    // Workspace layout
    char* ws = (char*)d_ws;
    float*    cbuf0 = (float*)ws;    ws += N * M * sizeof(float);            // leaf c (f32)
    _Float16* h16_0 = (_Float16*)ws; ws += N * M * sizeof(_Float16);         // leaf h (f16)
    float*    cP    = (float*)ws;    ws += (N / 2) * M * sizeof(float);      // ping c
    _Float16* hP    = (_Float16*)ws; ws += (N / 2) * M * sizeof(_Float16);   // ping h f16
    float*    cQ    = (float*)ws;    ws += (N / 4) * M * sizeof(float);      // pong c
    _Float16* hQ    = (_Float16*)ws; ws += (N / 4) * M * sizeof(_Float16);   // pong h f16
    _Float16* Bleaf = (_Float16*)ws; ws += (size_t)8  * 32 * 32 * 16 * sizeof(_Float16);
    _Float16* Btree = (_Float16*)ws; // 16*64*32*16 halfs

    // Pre-swizzle weights into WMMA B-fragment layout (tiny, L2-resident)
    k_swz_leaf<<<(8 * 32 * 32 * 16) / 256, 256, 0, stream>>>(Wcx, Wox, Bleaf);
    k_swz_tree<<<(16 * 64 * 32 * 16) / 256, 256, 0, stream>>>(Wl, Wr, Btree);

    // Leaf layer: h f32 -> d_out (the main output); h f16 -> level-1 A matrix.
    k_leaf<<<(unsigned)(N / 16), 256, 0, stream>>>(x, Bleaf, bcx, box,
                                                   out, h16_0, cbuf0);

    const _Float16* h_in = h16_0;
    const float*    c_in = cbuf0;
    for (int lvl = 1; (N >> lvl) >= 1; ++lvl) {
        const int n_out = (int)(N >> lvl);
        float*    c_o;
        _Float16* h16_o;
        float*    h32_o = nullptr;
        if (n_out == 1) {               // root -> tail of d_out: c_root, h_root
            c_o   = out + N * M;
            h32_o = out + N * M + M;
            h16_o = hP;                 // scratch; still written, unused
        } else if (lvl & 1) {
            c_o = cP; h16_o = hP;
        } else {
            c_o = cQ; h16_o = hQ;
        }
        const unsigned grid = (unsigned)((n_out + 15) / 16);
        k_level<<<grid, 256, 0, stream>>>(h_in, c_in, Btree, bl, br,
                                          c_o, h16_o, h32_o, n_out);
        h_in = h16_o; c_in = c_o;
    }
}
